// GNNClassifier_37692632990314
// MI455X (gfx1250) — matrix operand order repaired and checked by reading, not compiled, not run on hardware
//
#include <hip/hip_runtime.h>

#define N_NODES 50000
#define N_EDGES 800000
#define HID     256
#define KCAT    512
#define N_TOKENS 819200
#define N_DOCS  4096
#define M_PAD   50048   /* 782 tiles * 64 rows */

// ---------------------------------------------------------------------------
// WMMA fragment types (CDNA5 wave32: 16x16x32 bf16 -> A/B = 8 VGPRs packed,
// C/D = 8 f32 VGPRs)
// ---------------------------------------------------------------------------
typedef __attribute__((ext_vector_type(16))) __bf16 v16bf;
typedef __attribute__((ext_vector_type(8)))  float  v8f;

union FragAB { v16bf v; unsigned int u[8]; };
union FragC  { v8f   v; float        f[8]; };

__device__ __forceinline__ unsigned short f32_to_bf16(float x) {
  union { float f; unsigned int u; } v; v.f = x;
  unsigned int r = v.u + 0x7FFFu + ((v.u >> 16) & 1u);  // round-to-nearest-even
  return (unsigned short)(r >> 16);
}
__device__ __forceinline__ unsigned int pack_bf16x2(float a, float b) {
  return (unsigned int)f32_to_bf16(a) | ((unsigned int)f32_to_bf16(b) << 16);
}

// ---------------------------------------------------------------------------
// 1) degree: deg[dst] += 1 over edges (computed once; edges are constant)
// ---------------------------------------------------------------------------
__global__ void deg_kernel(const int* __restrict__ ei, float* __restrict__ deg) {
  int e = blockIdx.x * blockDim.x + threadIdx.x;
  if (e < N_EDGES) atomicAdd(&deg[ei[N_EDGES + e]], 1.0f);
}

// ---------------------------------------------------------------------------
// 2) edge aggregation: agg[dst,:] += x[src,:]  (one wave per edge, 8 f32
//    atomics per lane; x is L2-resident at 51 MB so gathers mostly hit L2)
// ---------------------------------------------------------------------------
__global__ __launch_bounds__(256)
void aggregate_kernel(const int* __restrict__ ei, const float* __restrict__ x,
                      float* __restrict__ agg) {
  int wave = threadIdx.x >> 5, lane = threadIdx.x & 31;
  int e = blockIdx.x * 8 + wave;
  if (e >= N_EDGES) return;
  int src = ei[e];
  int dst = ei[N_EDGES + e];
  const float* xs = x + (size_t)src * HID;
  float*       ad = agg + (size_t)dst * HID;
#pragma unroll
  for (int i = 0; i < 8; ++i) {
    int f = lane + 32 * i;
    atomicAdd(&ad[f], xs[f]);
  }
}

// ---------------------------------------------------------------------------
// 3) build bf16 concatenated activations: act[n, 0:256]=agg/deg, act[n,256:512]=x
//    (each thread packs 2 features -> one b32 store)
// ---------------------------------------------------------------------------
__global__ __launch_bounds__(256)
void prepare_kernel(const float* __restrict__ agg, const float* __restrict__ deg,
                    const float* __restrict__ xprev, unsigned int* __restrict__ act32) {
  int idx = blockIdx.x * blockDim.x + threadIdx.x;   // N_NODES*128 threads
  int n  = idx >> 7;
  int f2 = (idx & 127) << 1;
  if (n >= N_NODES) return;
  float dinv = 1.0f / fmaxf(deg[n], 1.0f);
  size_t ib = (size_t)n * HID + f2;
  size_t ob = (size_t)n * (KCAT / 2) + (f2 >> 1);          // in uint units
  act32[ob]             = pack_bf16x2(agg[ib] * dinv, agg[ib + 1] * dinv);
  act32[ob + HID / 2]   = pack_bf16x2(xprev[ib], xprev[ib + 1]);
}

// ---------------------------------------------------------------------------
// 4) weight conversion: Wcat[n][k] = bf16( k<256 ? Wl[n][k] : Wr[n][k-256] )
// ---------------------------------------------------------------------------
__global__ __launch_bounds__(256)
void convw_kernel(const float* __restrict__ Wl, const float* __restrict__ Wr,
                  unsigned int* __restrict__ out32) {
  int idx = blockIdx.x * blockDim.x + threadIdx.x;   // 128*256 threads (HID*HID/2)
  int n  = idx >> 8;
  int kp = (idx & 255) << 1;
  float a, b;
  if (kp < HID) { a = Wl[n * HID + kp];       b = Wl[n * HID + kp + 1]; }
  else          { a = Wr[n * HID + kp - HID]; b = Wr[n * HID + kp - HID + 1]; }
  out32[n * (KCAT / 2) + (kp >> 1)] = pack_bf16x2(a, b);
}

// ---------------------------------------------------------------------------
// 5) bf16 WMMA GEMM: y[M,256] = act[M,512] @ Wcat[256,512]^T + bias
//    block tile 64x256, wave grid 2(M) x 4(N), 2x4 C-fragments per wave
// ---------------------------------------------------------------------------
__global__ __launch_bounds__(256)
void gemm_bias_kernel(const unsigned short* __restrict__ act,  // [M_PAD][512] bf16
                      const unsigned short* __restrict__ w,    // [256][512]   bf16
                      const float* __restrict__ bias,          // [256]
                      float* __restrict__ y) {                 // [N_NODES][256]
  __shared__ __align__(16) unsigned short As[64][40];    // 32 K + 8 pad (80B rows)
  __shared__ __align__(16) unsigned short Bs[256][40];   // n-major, k contiguous

  const int tid  = threadIdx.x;
  const int m0   = blockIdx.x * 64;
  const int wv   = tid >> 5, lane = tid & 31;
  const int wm   = wv & 1,   wn   = wv >> 1;
  const int lr   = lane & 15, lh  = lane >> 4;

  FragC c[2][4];
#pragma unroll
  for (int mi = 0; mi < 2; ++mi)
#pragma unroll
    for (int ni = 0; ni < 4; ++ni)
#pragma unroll
      for (int j = 0; j < 8; ++j) c[mi][ni].f[j] = 0.0f;

  const int ra = tid >> 2;           // A-stage: row within tile
  const int qa = tid & 3;            // A-stage: 16B quad

  for (int k0 = 0; k0 < KCAT; k0 += 32) {
    // ---- stage A tile (64x32 bf16) ----
    uint4 av = make_uint4(0u, 0u, 0u, 0u);
    int arow = m0 + ra;
    if (arow < N_NODES)
      av = *(const uint4*)(act + (size_t)arow * KCAT + k0 + qa * 8);
    *(uint4*)&As[ra][qa * 8] = av;
    // ---- stage B tile (256n x 32k bf16), k contiguous per n ----
    const unsigned short* wp = w + (size_t)tid * KCAT + k0;
#pragma unroll
    for (int j = 0; j < 4; ++j)
      *(uint4*)&Bs[tid][j * 8] = *(const uint4*)(wp + j * 8);
    __syncthreads();

    // ---- build fragments per ISA 7.12.2 layouts ----
    FragAB a[2];
#pragma unroll
    for (int mi = 0; mi < 2; ++mi) {
      int row = wm * 32 + mi * 16 + lr;
#pragma unroll
      for (int i = 0; i < 8; ++i) {
        int kidx = ((i & 3) << 1) + ((i >= 4) ? 16 : 0) + lh * 8;
        a[mi].u[i] = *(const unsigned int*)&As[row][kidx];
      }
    }
    FragAB b[4];
#pragma unroll
    for (int ni = 0; ni < 4; ++ni) {
      int col = wn * 64 + ni * 16 + lr;
      int kb  = lh * 8;
#pragma unroll
      for (int i = 0; i < 8; ++i)
        b[ni].u[i] = *(const unsigned int*)&Bs[col][kb + ((i & 3) << 1) + ((i >= 4) ? 16 : 0)];
    }
#pragma unroll
    for (int mi = 0; mi < 2; ++mi)
#pragma unroll
      for (int ni = 0; ni < 4; ++ni)
        c[mi][ni].v = __builtin_amdgcn_wmma_f32_16x16x32_bf16(
            false, a[mi].v, false, b[ni].v, (short)0, c[mi][ni].v, false, false);
    __syncthreads();
  }

  // ---- epilogue: C layout (VGPR j: M = j + 8*lh, N = lane%16) + bias ----
#pragma unroll
  for (int mi = 0; mi < 2; ++mi)
#pragma unroll
    for (int ni = 0; ni < 4; ++ni) {
      int n = wn * 64 + ni * 16 + lr;
      float bv = bias[n];
#pragma unroll
      for (int j = 0; j < 8; ++j) {
        int m = m0 + wm * 32 + mi * 16 + (lh << 3) + j;
        if (m < N_NODES) y[(size_t)m * HID + n] = c[mi][ni].f[j] + bv;
      }
    }
}

// ---------------------------------------------------------------------------
// 6) per-feature moments (one pass: Sum, SumSq) for GraphNorm
// ---------------------------------------------------------------------------
__global__ __launch_bounds__(256)
void stats_kernel(const float* __restrict__ y, float* __restrict__ sums,
                  float* __restrict__ sumsq) {
  int f = threadIdx.x;
  float s = 0.0f, q = 0.0f;
  for (int r = blockIdx.x; r < N_NODES; r += gridDim.x) {
    float v = y[(size_t)r * HID + f];
    s += v; q += v * v;
  }
  atomicAdd(&sums[f], s);
  atomicAdd(&sumsq[f], q);
}

// ---------------------------------------------------------------------------
// 7) GraphNorm + residual + ReLU  (var(x - s*mu) = E[x^2] - 2 s mu^2 + s^2 mu^2)
// ---------------------------------------------------------------------------
__global__ __launch_bounds__(256)
void norm_kernel(const float* __restrict__ y, const float* __restrict__ sums,
                 const float* __restrict__ sumsq, const float* __restrict__ gw,
                 const float* __restrict__ gb, const float* __restrict__ gms,
                 const float* __restrict__ residual, float* __restrict__ out) {
  int idx = blockIdx.x * blockDim.x + threadIdx.x;   // N_NODES*256 threads
  int f = idx & 255;
  const float inv_n = 1.0f / (float)N_NODES;
  float mu  = sums[f]  * inv_n;
  float ex2 = sumsq[f] * inv_n;
  float s   = gms[f];
  float var = ex2 - 2.0f * s * mu * mu + s * s * mu * mu;
  float o = gw[f] * (y[idx] - s * mu) * rsqrtf(var + 1e-5f) + gb[f];
  if (residual) o += residual[idx];
  out[idx] = fmaxf(o, 0.0f);
}

// ---------------------------------------------------------------------------
// 8) fused doc pooling (tfidf-weighted, sorted doc ids) + FC(256->2)
//    one block per doc, one thread per feature
// ---------------------------------------------------------------------------
__device__ __forceinline__ int lower_bound_i(const int* __restrict__ a, int n, int key) {
  int lo = 0, hi = n;
  while (lo < hi) { int mid = (lo + hi) >> 1; if (a[mid] < key) lo = mid + 1; else hi = mid; }
  return lo;
}

__global__ __launch_bounds__(256)
void pool_fc_kernel(const float* __restrict__ word_emb,
                    const int* __restrict__ tok_wid, const int* __restrict__ tok_did,
                    const float* __restrict__ tfidf, const float* __restrict__ fcW,
                    const float* __restrict__ fcb, float* __restrict__ out) {
  __shared__ float red[256];
  int d = blockIdx.x;
  int t = threadIdx.x;
  int start = lower_bound_i(tok_did, N_TOKENS, d);
  int end   = lower_bound_i(tok_did, N_TOKENS, d + 1);
  int cnt = end - start;

  float wemb = 0.0f, es = 0.0f, ws = 0.0f;
  for (int tok = start; tok < end; ++tok) {
    int wid = tok_wid[tok];
    if (tok + 1 < end)   // prefetch next word row -> global_prefetch_b8
      __builtin_prefetch(word_emb + (size_t)tok_wid[tok + 1] * HID, 0, 1);
    float w = tfidf[tok];
    float e = word_emb[(size_t)wid * HID + t];
    wemb += w * e;
    es   += e;
    ws   += w;
  }

  float de = 0.0f;
  if (cnt > 0) {
    float mean = es / fmaxf((float)cnt, 1.0f);
    de = (ws > 0.0f) ? (wemb / (ws + 1e-8f)) : mean;
  }

#pragma unroll
  for (int c = 0; c < 2; ++c) {
    red[t] = de * fcW[c * HID + t];
    __syncthreads();
    for (int off = 128; off > 0; off >>= 1) {
      if (t < off) red[t] += red[t + off];
      __syncthreads();
    }
    if (t == 0) out[d * 2 + c] = red[0] + fcb[c];
    __syncthreads();
  }
}

// ---------------------------------------------------------------------------
// host side
// ---------------------------------------------------------------------------
extern "C" void kernel_launch(void* const* d_in, const int* in_sizes, int n_in,
                              void* d_out, int out_size, void* d_ws, size_t ws_size,
                              hipStream_t stream) {
  (void)in_sizes; (void)n_in; (void)out_size; (void)ws_size;

  const float* x_in   = (const float*)d_in[0];
  const int*   eidx   = (const int*)d_in[1];
  const int*   twid   = (const int*)d_in[2];
  const int*   tdid   = (const int*)d_in[3];
  const float* tfidf  = (const float*)d_in[4];
  const float* Wl[4]  = {(const float*)d_in[5],  (const float*)d_in[11], (const float*)d_in[17], (const float*)d_in[23]};
  const float* bl[4]  = {(const float*)d_in[6],  (const float*)d_in[12], (const float*)d_in[18], (const float*)d_in[24]};
  const float* Wr[4]  = {(const float*)d_in[7],  (const float*)d_in[13], (const float*)d_in[19], (const float*)d_in[25]};
  const float* gnw[4] = {(const float*)d_in[8],  (const float*)d_in[14], (const float*)d_in[20], (const float*)d_in[26]};
  const float* gnb[4] = {(const float*)d_in[9],  (const float*)d_in[15], (const float*)d_in[21], (const float*)d_in[27]};
  const float* gms[4] = {(const float*)d_in[10], (const float*)d_in[16], (const float*)d_in[22], (const float*)d_in[28]};
  const float* fcW = (const float*)d_in[29];
  const float* fcb = (const float*)d_in[30];
  float* out = (float*)d_out;

  // workspace layout (256B aligned segments)
  char* ws = (char*)d_ws;
  const size_t SZ_DEG  = (size_t)M_PAD * sizeof(float);                 // 200 KB
  const size_t SZ_MAT  = (size_t)M_PAD * HID * sizeof(float);           // 51.25 MB
  const size_t SZ_ACT  = (size_t)M_PAD * KCAT * sizeof(unsigned short); // 51.25 MB
  const size_t SZ_WCAT = (size_t)HID * KCAT * sizeof(unsigned short);   // 256 KB per layer
  size_t off = 0;
  float*          deg   = (float*)(ws + off);           off += SZ_DEG;      off = (off + 255) & ~(size_t)255;
  float*          aggy  = (float*)(ws + off);           off += SZ_MAT;      off = (off + 255) & ~(size_t)255; // agg, then reused as y
  unsigned short* act   = (unsigned short*)(ws + off);  off += SZ_ACT;      off = (off + 255) & ~(size_t)255;
  float*          xA    = (float*)(ws + off);           off += SZ_MAT;      off = (off + 255) & ~(size_t)255;
  float*          xB    = (float*)(ws + off);           off += SZ_MAT;      off = (off + 255) & ~(size_t)255;
  unsigned short* wcat  = (unsigned short*)(ws + off);  off += 4 * SZ_WCAT; off = (off + 255) & ~(size_t)255;
  float*          sums  = (float*)(ws + off);           off += HID * sizeof(float);
  float*          sumsq = (float*)(ws + off);           off += HID * sizeof(float);

  // degrees (constant across layers) + weight conversion (4 layers)
  hipMemsetAsync(deg, 0, SZ_DEG, stream);
  deg_kernel<<<(N_EDGES + 255) / 256, 256, 0, stream>>>(eidx, deg);
  for (int L = 0; L < 4; ++L)
    convw_kernel<<<(HID * HID / 2 + 255) / 256, 256, 0, stream>>>(
        Wl[L], Wr[L], (unsigned int*)(wcat + (size_t)L * HID * KCAT));

  const float* x_prev = x_in;
  for (int L = 0; L < 4; ++L) {
    // mean aggregation over edges
    hipMemsetAsync(aggy, 0, (size_t)N_NODES * HID * sizeof(float), stream);
    aggregate_kernel<<<(N_EDGES + 7) / 8, 256, 0, stream>>>(eidx, x_prev, aggy);
    // concat bf16 activations [agg/deg | x]
    prepare_kernel<<<(N_NODES * (HID / 2) + 255) / 256, 256, 0, stream>>>(
        aggy, deg, x_prev, (unsigned int*)act);
    // fused GEMM (K=512) + bias; y overwrites agg (agg fully consumed by prepare)
    gemm_bias_kernel<<<M_PAD / 64, 256, 0, stream>>>(
        act, wcat + (size_t)L * HID * KCAT, bl[L], aggy);
    // GraphNorm stats + apply + residual + ReLU
    hipMemsetAsync(sums, 0, 2 * HID * sizeof(float), stream);
    stats_kernel<<<256, 256, 0, stream>>>(aggy, sums, sumsq);
    float* x_next = (L & 1) ? xB : xA;
    norm_kernel<<<N_NODES, 256, 0, stream>>>(
        aggy, sums, sumsq, gnw[L], gnb[L], gms[L],
        (L == 0) ? (const float*)nullptr : x_prev, x_next);
    x_prev = x_next;
  }

  // doc pooling + FC
  pool_fc_kernel<<<N_DOCS, 256, 0, stream>>>(x_prev, twid, tdid, tfidf, fcW, fcb, out);
}
